// RoPEAttention_17497696764408
// MI455X (gfx1250) — matrix operand order repaired and checked by reading, not compiled, not run on hardware
//
#include <hip/hip_runtime.h>
#include <math.h>

#define DIM     768
#define NHEADS  12
#define HD      64
#define NTOK    1025
#define BATCH   16
#define BH      (BATCH * NHEADS)   // 192
#define MROWS   (BATCH * NTOK)     // 16400
#define QKVDIM  (3 * DIM)          // 2304

typedef __attribute__((ext_vector_type(16))) __bf16 v16bf;
typedef __attribute__((ext_vector_type(8)))  __bf16 v8bf;
typedef __attribute__((ext_vector_type(8)))  float  v8f;

union FragB { v16bf v; v8bf h[2]; __bf16 e[16]; };
union V8U   { v8bf  v; __bf16 e[8]; unsigned short u[8]; };

__device__ __forceinline__ v8f wmma_bf16(v16bf a, v16bf b, v8f c) {
  // 8 args: (neg_a, A, neg_b, B, c_mod, C, reuse_a, reuse_b)
  return __builtin_amdgcn_wmma_f32_16x16x32_bf16(false, a, false, b, (short)0, c,
                                                 false, false);
}

__device__ __forceinline__ unsigned short bf_bits(__bf16 x) {
  union { __bf16 b; unsigned short u; } c;
  c.b = x;
  return c.u;
}

// ---------------------------------------------------------------------------
// Stage 0: fp32 -> bf16 conversion (grid-stride)
// ---------------------------------------------------------------------------
__global__ void cvt_f32_bf16(const float* __restrict__ src,
                             __bf16* __restrict__ dst, int n) {
  int i = blockIdx.x * blockDim.x + threadIdx.x;
  int stride = gridDim.x * blockDim.x;
  for (; i < n; i += stride) dst[i] = (__bf16)src[i];
}

// ---------------------------------------------------------------------------
// Tiled WMMA GEMM: C[M x Nn] = A[M x K] * W[Nn x K]^T (+ bias)
// Block = 256 threads = 8 waves; block tile 128x128; wave tile 64x32.
// Branch-free staging: OOB A rows are CLAMPED (their C rows are never stored).
// ---------------------------------------------------------------------------
#define TILE_M 128
#define TILE_N 128
#define TILE_K 32
#define LDA_S  40   // 32 + 8 pad (bf16 elems); 80B rows keep 16B alignment

template <bool BF16OUT, bool HASBIAS>
__global__ __launch_bounds__(256)
void gemm_bf16_wmma(const __bf16* __restrict__ A,
                    const __bf16* __restrict__ W,
                    void* __restrict__ Cout,
                    const float* __restrict__ bias,
                    int M, int Nn, int K) {
  __shared__ __bf16 As[TILE_M * LDA_S];
  __shared__ __bf16 Ws[TILE_N * LDA_S];

  const int tid  = threadIdx.x;
  const int wave = tid >> 5;
  const int lane = tid & 31;
  const int half = lane >> 4;
  const int ln   = lane & 15;
  const int m0   = blockIdx.y * TILE_M;
  const int n0   = blockIdx.x * TILE_N;
  const int wm0  = (wave & 1) * 64;   // 2 waves along M
  const int wn0  = (wave >> 1) * 32;  // 4 waves along N

  v8f acc[4][2];
  {
    v8f z = {};
#pragma unroll
    for (int tm = 0; tm < 4; ++tm)
#pragma unroll
      for (int tn = 0; tn < 2; ++tn) acc[tm][tn] = z;
  }

  const int lrow = tid >> 1;            // 0..127
  const int lseg = (tid & 1) * 16;      // 0 or 16 (elems)

  // clamp A row (branch-free; OOB rows only feed OOB C rows)
  int gm = m0 + lrow;
  if (gm >= M) gm = M - 1;
  const __bf16* pa = A + (size_t)gm * K + lseg;
  const __bf16* pw = W + (size_t)(n0 + lrow) * K + lseg;

  for (int k0 = 0; k0 < K; k0 += TILE_K) {
    // ---- stage A tile (128 x 32) + W tile (128 x 32), branch-free ----
    v8bf a0 = *(const v8bf*)(pa + k0);
    v8bf a1 = *(const v8bf*)(pa + k0 + 8);
    v8bf w0 = *(const v8bf*)(pw + k0);
    v8bf w1 = *(const v8bf*)(pw + k0 + 8);
    __builtin_prefetch(pa + k0 + TILE_K, 0, 1);   // speculative, safe past end
    __builtin_prefetch(pw + k0 + TILE_K, 0, 1);
    *(v8bf*)&As[lrow * LDA_S + lseg]     = a0;
    *(v8bf*)&As[lrow * LDA_S + lseg + 8] = a1;
    *(v8bf*)&Ws[lrow * LDA_S + lseg]     = w0;
    *(v8bf*)&Ws[lrow * LDA_S + lseg + 8] = w1;
    __syncthreads();

    // ---- load ALL fragments first (single dscnt wait), then 8 WMMAs ----
    FragB bf[2], af[4];
#pragma unroll
    for (int tn = 0; tn < 2; ++tn) {
      const __bf16* p = &Ws[(wn0 + tn * 16 + ln) * LDA_S + 16 * half];
      bf[tn].h[0] = *(const v8bf*)p;
      bf[tn].h[1] = *(const v8bf*)(p + 8);
    }
#pragma unroll
    for (int tm = 0; tm < 4; ++tm) {
      const __bf16* p = &As[(wm0 + tm * 16 + ln) * LDA_S];
      af[tm].h[0] = *(const v8bf*)(p + 16 * half);
      af[tm].h[1] = *(const v8bf*)(p + 16 * half + 8);
    }
#pragma unroll
    for (int tm = 0; tm < 4; ++tm)
#pragma unroll
      for (int tn = 0; tn < 2; ++tn)
        acc[tm][tn] = wmma_bf16(af[tm].v, bf[tn].v, acc[tm][tn]);
    __syncthreads();
  }

  // ---- single straight-line epilogue (templated: no dual paths) ----
#pragma unroll
  for (int tm = 0; tm < 4; ++tm)
#pragma unroll
    for (int tn = 0; tn < 2; ++tn) {
      int n = n0 + wn0 + tn * 16 + ln;
      float bv = HASBIAS ? bias[n] : 0.f;
#pragma unroll
      for (int r = 0; r < 8; ++r) {
        int m = m0 + wm0 + tm * 16 + r + 8 * half;
        if (m < M) {
          float v = acc[tm][tn][r] + bv;
          if (BF16OUT) ((__bf16*)Cout)[(size_t)m * Nn + n] = (__bf16)v;
          else         ((float*)Cout)[(size_t)m * Nn + n] = v;
        }
      }
    }
}

// ---------------------------------------------------------------------------
// RoPE + head split: qkv (B*N, 2304) bf16 -> Q,K,V (B*H, N, 64) bf16.
// Q pre-scaled by HD^-0.5; CLS token (n==0) unrotated.
// ---------------------------------------------------------------------------
__global__ void rope_split_kernel(const __bf16* __restrict__ qkv,
                                  const float* __restrict__ freqs,
                                  __bf16* __restrict__ Qo,
                                  __bf16* __restrict__ Ko,
                                  __bf16* __restrict__ Vo) {
  int idx = blockIdx.x * blockDim.x + threadIdx.x;
  if (idx >= MROWS * NHEADS) return;
  int h  = idx % NHEADS;
  int bn = idx / NHEADS;
  int n  = bn % NTOK;
  int b  = bn / NTOK;
  const __bf16* src = qkv + (size_t)bn * QKVDIM + h * HD;
  size_t dst = ((size_t)(b * NHEADS + h) * NTOK + n) * HD;
  const float scale = 0.125f;  // 64^-0.5
  if (n == 0) {
    for (int d = 0; d < HD; ++d) {
      Qo[dst + d] = (__bf16)((float)src[d] * scale);
      Ko[dst + d] = src[DIM + d];
      Vo[dst + d] = src[2 * DIM + d];
    }
    return;
  }
  int t = n - 1;
  float tx = (float)(t & 31);
  float ty = (float)(t >> 5);
  for (int i = 0; i < HD / 2; ++i) {
    float ang = tx * freqs[h * 32 + i] + ty * freqs[384 + h * 32 + i];
    float sa, ca;
    __sincosf(ang, &sa, &ca);
    float q0 = (float)src[2 * i],       q1 = (float)src[2 * i + 1];
    float k0 = (float)src[DIM + 2 * i], k1 = (float)src[DIM + 2 * i + 1];
    Qo[dst + 2 * i]     = (__bf16)((q0 * ca - q1 * sa) * scale);
    Qo[dst + 2 * i + 1] = (__bf16)((q0 * sa + q1 * ca) * scale);
    Ko[dst + 2 * i]     = (__bf16)(k0 * ca - k1 * sa);
    Ko[dst + 2 * i + 1] = (__bf16)(k0 * sa + k1 * ca);
    Vo[dst + 2 * i]     = src[2 * DIM + 2 * i];
    Vo[dst + 2 * i + 1] = src[2 * DIM + 2 * i + 1];
  }
}

// ---------------------------------------------------------------------------
// Flash attention: block = (b,h) x 128-query tile; 8 waves x 16 queries.
// Scores never touch HBM: S in accumulators, P via per-wave LDS.
//
// The inner k-dimension of P*V is PERMUTED: original token col (t*16+ln) is
// stored at k' = ln*8 + t in both Ps (columns) and Vs (rows). P*V is invariant
// under any shared k-permutation, and this makes every lane's 8 P values per
// row contiguous (one ds_store_b128) and V staging dword-packable
// (conflict-free ds_store_b32), instead of bank-serialized b16 scatters.
// ---------------------------------------------------------------------------
__global__ __launch_bounds__(256)
void attn_fa_wmma(const __bf16* __restrict__ Qg, const __bf16* __restrict__ Kg,
                  const __bf16* __restrict__ Vg, __bf16* __restrict__ Og) {
  __shared__ __bf16 Ks[128 * 72];       // (token, hd), stride 72
  __shared__ __bf16 Vs[HD * 136];       // (hd, k'), stride 136
  __shared__ __bf16 Ps[8][16 * 136];    // per-wave P tile 16 x 128 (cols = k')

  const int tid  = threadIdx.x;
  const int wave = tid >> 5;
  const int lane = tid & 31;
  const int half = lane >> 4;
  const int ln   = lane & 15;
  const int q0   = blockIdx.x * 128;
  const int bh   = blockIdx.y;
  const int b    = bh / NHEADS;
  const int h    = bh % NHEADS;

  // Q A-fragments straight from global (rows = lane%16, K split by lane/16);
  // OOB query rows clamped -- their outputs are never stored.
  FragB aq[2];
  {
    int qrow = q0 + wave * 16 + ln;
    if (qrow >= NTOK) qrow = NTOK - 1;
    const __bf16* p = Qg + ((size_t)bh * NTOK + qrow) * HD;
#pragma unroll
    for (int ks = 0; ks < 2; ++ks) {
      aq[ks].h[0] = *(const v8bf*)(p + ks * 32 + 16 * half);
      aq[ks].h[1] = *(const v8bf*)(p + ks * 32 + 16 * half + 8);
    }
  }

  float mi[8], li[8];
  v8f o[4];
  {
    v8f z = {};
#pragma unroll
    for (int r = 0; r < 8; ++r) { mi[r] = -3.0e38f; li[r] = 0.f; }
#pragma unroll
    for (int t = 0; t < 4; ++t) o[t] = z;
  }

  // K staging map: thread -> (token row, 32-elem hd segment)
  const int krow = tid >> 1;         // 0..127
  const int kseg = (tid & 1) * 32;   // 0 or 32

  // V staging map: thread -> (token pair sharing a k' dword, 16-elem hd seg)
  const int vln  = tid & 15;          // token ln
  const int vtp  = (tid >> 4) & 3;    // token t-pair: t0 = 2*vtp
  const int vds  = (tid >> 6) * 16;   // hd segment: 0,16,32,48
  const int vkp  = vln * 8 + 2 * vtp; // k' of first token of the pair

  const size_t kvbase = (size_t)bh * NTOK;

  for (int j = 0; j < (NTOK + 127) / 128; ++j) {
    const int kv0 = j * 128;
    // ---- stage K (token, hd) row-major, b128 stores ----
    {
      int tok = kv0 + krow;
      if (tok >= NTOK) tok = NTOK - 1;
      const __bf16* kp = Kg + (kvbase + tok) * HD + kseg;
      v8bf k0 = *(const v8bf*)kp;
      v8bf k1 = *(const v8bf*)(kp + 8);
      v8bf k2 = *(const v8bf*)(kp + 16);
      v8bf k3 = *(const v8bf*)(kp + 24);
      *(v8bf*)&Ks[krow * 72 + kseg]      = k0;
      *(v8bf*)&Ks[krow * 72 + kseg + 8]  = k1;
      *(v8bf*)&Ks[krow * 72 + kseg + 16] = k2;
      *(v8bf*)&Ks[krow * 72 + kseg + 24] = k3;
    }
    // ---- stage V transposed+permuted: Vs[d][k'] via packed b32 stores ----
    {
      int tokA = kv0 + (2 * vtp) * 16 + vln;       // maps to k' = vkp
      int tokB = tokA + 16;                        // maps to k' = vkp + 1
      if (tokA >= NTOK) tokA = NTOK - 1;
      if (tokB >= NTOK) tokB = NTOK - 1;
      const __bf16* pA = Vg + (kvbase + tokA) * HD + vds;
      const __bf16* pB = Vg + (kvbase + tokB) * HD + vds;
      V8U a0, a1, b0, b1;
      a0.v = *(const v8bf*)pA;  a1.v = *(const v8bf*)(pA + 8);
      b0.v = *(const v8bf*)pB;  b1.v = *(const v8bf*)(pB + 8);
#pragma unroll
      for (int i = 0; i < 8; ++i) {
        unsigned p0 = (unsigned)a0.u[i] | ((unsigned)b0.u[i] << 16);
        unsigned p1 = (unsigned)a1.u[i] | ((unsigned)b1.u[i] << 16);
        *(unsigned*)&Vs[(vds + i) * 136 + vkp]     = p0;
        *(unsigned*)&Vs[(vds + 8 + i) * 136 + vkp] = p1;
      }
    }
    __syncthreads();

    // ---- S = Q K^T (16 x 128) in fp32 accumulators ----
    v8f s[8];
#pragma unroll
    for (int t = 0; t < 8; ++t) {
      v8f z = {};
      s[t] = z;
#pragma unroll
      for (int ks = 0; ks < 2; ++ks) {
        FragB bk;
        const __bf16* p = &Ks[(t * 16 + ln) * 72 + ks * 32 + 16 * half];
        bk.h[0] = *(const v8bf*)p;
        bk.h[1] = *(const v8bf*)(p + 8);
        s[t] = wmma_bf16(aq[ks].v, bk.v, s[t]);
      }
      if (kv0 + t * 16 + ln >= NTOK)
#pragma unroll
        for (int r = 0; r < 8; ++r) s[t][r] = -1.0e30f;
    }

    // ---- online softmax; P row chunk packed to one b128 store per row ----
#pragma unroll
    for (int r = 0; r < 8; ++r) {
      float mx = -3.0e38f;
#pragma unroll
      for (int t = 0; t < 8; ++t) mx = fmaxf(mx, s[t][r]);
#pragma unroll
      for (int msk = 1; msk < 16; msk <<= 1)
        mx = fmaxf(mx, __shfl_xor(mx, msk, 32));
      float mnew = fmaxf(mi[r], mx);
      float alpha = __expf(mi[r] - mnew);
      mi[r] = mnew;
      li[r] *= alpha;
#pragma unroll
      for (int t = 0; t < 4; ++t) o[t][r] *= alpha;
      float rs = 0.f;
      V8U pw;
#pragma unroll
      for (int t = 0; t < 8; ++t) {
        float pv = __expf(s[t][r] - mnew);
        rs += pv;
        pw.e[t] = (__bf16)pv;         // col' = ln*8 + t : contiguous per lane
      }
      *(v8bf*)&Ps[wave][(r + 8 * half) * 136 + ln * 8] = pw.v;
#pragma unroll
      for (int msk = 1; msk < 16; msk <<= 1) rs += __shfl_xor(rs, msk, 32);
      li[r] += rs;
    }

    // ---- O += P V  over permuted k' (4 wmma k-steps) ----
#pragma unroll
    for (int ks = 0; ks < 4; ++ks) {
      FragB ap;
      const __bf16* p = &Ps[wave][ln * 136 + ks * 32];
      ap.h[0] = *(const v8bf*)(p + 16 * half);
      ap.h[1] = *(const v8bf*)(p + 16 * half + 8);
#pragma unroll
      for (int tn = 0; tn < 4; ++tn) {
        FragB bv;
        const __bf16* vp = &Vs[(tn * 16 + ln) * 136 + ks * 32 + 16 * half];
        bv.h[0] = *(const v8bf*)vp;
        bv.h[1] = *(const v8bf*)(vp + 8);
        o[tn] = wmma_bf16(ap.v, bv.v, o[tn]);
      }
    }
    __syncthreads();
  }

  // ---- normalize + write (b, n, h*64+d) bf16 ----
#pragma unroll
  for (int r = 0; r < 8; ++r) {
    int q = q0 + wave * 16 + r + 8 * half;
    if (q >= NTOK) continue;
    float inv = (li[r] > 0.f) ? (1.f / li[r]) : 0.f;
    __bf16* op = Og + ((size_t)(b * NTOK + q)) * DIM + h * HD;
#pragma unroll
    for (int tn = 0; tn < 4; ++tn)
      op[tn * 16 + ln] = (__bf16)(o[tn][r] * inv);
  }
}

// ---------------------------------------------------------------------------
extern "C" void kernel_launch(void* const* d_in, const int* in_sizes, int n_in,
                              void* d_out, int out_size, void* d_ws, size_t ws_size,
                              hipStream_t stream) {
  (void)in_sizes; (void)n_in; (void)out_size; (void)ws_size;
  const float* x     = (const float*)d_in[0];
  const float* Wqkv  = (const float*)d_in[1];
  const float* Wproj = (const float*)d_in[2];
  const float* bproj = (const float*)d_in[3];
  const float* freqs = (const float*)d_in[4];

  char* ws = (char*)d_ws;
  size_t off = 0;
  auto alloc = [&](size_t bytes) -> void* {
    void* p = ws + off;
    off += (bytes + 255) & ~(size_t)255;
    return p;
  };
  __bf16* x_bf     = (__bf16*)alloc((size_t)MROWS * DIM * 2);
  __bf16* wqkv_bf  = (__bf16*)alloc((size_t)QKVDIM * DIM * 2);
  __bf16* wproj_bf = (__bf16*)alloc((size_t)DIM * DIM * 2);
  __bf16* qkv_bf   = (__bf16*)alloc((size_t)MROWS * QKVDIM * 2);
  __bf16* Qbf      = (__bf16*)alloc((size_t)BH * NTOK * HD * 2);
  __bf16* Kbf      = (__bf16*)alloc((size_t)BH * NTOK * HD * 2);
  __bf16* Vbf      = (__bf16*)alloc((size_t)BH * NTOK * HD * 2);
  __bf16* attn_bf  = (__bf16*)alloc((size_t)MROWS * DIM * 2);

  // 0) fp32 -> bf16
  cvt_f32_bf16<<<2048, 256, 0, stream>>>(x,     x_bf,     MROWS * DIM);
  cvt_f32_bf16<<<512,  256, 0, stream>>>(Wqkv,  wqkv_bf,  QKVDIM * DIM);
  cvt_f32_bf16<<<256,  256, 0, stream>>>(Wproj, wproj_bf, DIM * DIM);

  // 1) QKV GEMM: (16400 x 768) x (2304 x 768)^T -> bf16
  gemm_bf16_wmma<true, false>
      <<<dim3(QKVDIM / TILE_N, (MROWS + TILE_M - 1) / TILE_M), 256, 0,
         stream>>>(x_bf, wqkv_bf, qkv_bf, nullptr, MROWS, QKVDIM, DIM);

  // 2) RoPE + head split (Q pre-scaled)
  rope_split_kernel<<<(MROWS * NHEADS + 255) / 256, 256, 0, stream>>>(
      qkv_bf, freqs, Qbf, Kbf, Vbf);

  // 3) Flash attention
  attn_fa_wmma<<<dim3((NTOK + 127) / 128, BH), 256, 0, stream>>>(
      Qbf, Kbf, Vbf, attn_bf);

  // 4) Output projection + bias -> fp32 d_out
  gemm_bf16_wmma<false, true>
      <<<dim3(DIM / TILE_N, (MROWS + TILE_M - 1) / TILE_M), 256, 0,
         stream>>>(attn_bf, wproj_bf, d_out, bproj, MROWS, DIM, DIM);
}
